// FixedAttention_85830626443606
// MI455X (gfx1250) — matrix operand order repaired and checked by reading, not compiled, not run on hardware
//
#include <hip/hip_runtime.h>
#include <hip/hip_bf16.h>
#include <math.h>

// ---------------------------------------------------------------------------
// Fused causal attention (fixed att scores) for MI455X / gfx1250, wave32.
//   K1: v = x @ W_attn^T + b_attn      -> bf16, swizzled into WMMA-B layout
//   K2: flash-style online softmax over causal att + P@V via wmma bf16
//   K3: out = y @ W_proj^T + b_proj    -> f32
// B=4, T=2048, C=512, NH=8, HD=64.
// Roofline: att lower-triangle ~256 MB @ 23.3 TB/s ~= 11 us; all matmul FLOPs
// (~26 GFLOP) are negligible at WMMA rates => K2 must stay pure-streaming.
// ---------------------------------------------------------------------------

typedef __bf16 bf16_t;
typedef __attribute__((ext_vector_type(16))) __bf16 v16bf;
typedef __attribute__((ext_vector_type(8)))  float  v8f;

#define TSEQ 2048
#define CDIM 512
#define NHEAD 8
#define HDIM 64

__device__ __forceinline__ float4 ldg4(const float* p) {
    return *reinterpret_cast<const float4*>(p);
}

// pack 8 f32 -> bf16 into elements [base .. base+7] of a v16bf
__device__ __forceinline__ void cvt8(v16bf& d, const int base,
                                     const float4 a, const float4 b) {
    d[base + 0] = (bf16_t)a.x; d[base + 1] = (bf16_t)a.y;
    d[base + 2] = (bf16_t)a.z; d[base + 3] = (bf16_t)a.w;
    d[base + 4] = (bf16_t)b.x; d[base + 5] = (bf16_t)b.y;
    d[base + 6] = (bf16_t)b.z; d[base + 7] = (bf16_t)b.w;
}

__device__ __forceinline__ v8f wmma_bf16(v16bf a, v16bf b, v8f c) {
    // D = A(16x32 bf16) * B(32x16 bf16) + C(16x16 f32)
    return __builtin_amdgcn_wmma_f32_16x16x32_bf16(
        /*neg_a=*/false, a, /*neg_b=*/false, b,
        /*c_mod=*/(short)0, c, /*reuse_a=*/false, /*reuse_b=*/false);
}

union Pack8 { bf16_t h[8]; uint4 u; };

// ---------------------------------------------------------------------------
// Kernel 1: v-projection. One wave computes a 16(token) x 64(one head) tile.
// All 20 raw f32 tile loads are issued before any conversion so the memory
// pipe keeps many b128 loads in flight (x and W are L2-resident).
// Output stored bf16 into vws pre-swizzled so K2 can load WMMA B tiles
// directly: tile(bh, kc, dt) = 512 bf16; lane L owns elems [L*16 .. L*16+15].
// ---------------------------------------------------------------------------
__global__ void __launch_bounds__(128) k_vproj(const float* __restrict__ x,
                                               const float* __restrict__ W,
                                               const float* __restrict__ bias,
                                               bf16_t* __restrict__ vws)
{
    const int lane  = threadIdx.x & 31;
    const int wid   = blockIdx.x * 4 + (threadIdx.x >> 5);   // 0..4095
    const int mtile = wid >> 3;                              // 0..511
    const int h     = wid & 7;
    const int mbase = mtile << 4;
    const int rlo   = lane & 15;
    const bool hi   = lane >= 16;
    const int koffA = hi ? 8 : 0;
    const int koffB = hi ? 8 : 0;

    const float* xrow = x + (size_t)(mbase + rlo) * CDIM;
    const float* wr[4];
    #pragma unroll
    for (int dt = 0; dt < 4; ++dt)
        wr[dt] = W + (size_t)(h * HDIM + dt * 16 + rlo) * CDIM;

    v8f acc[4] = {};

    for (int kc = 0; kc < CDIM; kc += 32) {
        // ---- issue all raw loads first (20 independent b128) ----
        float4 ar[4];
        ar[0] = ldg4(xrow + kc + koffA);
        ar[1] = ldg4(xrow + kc + koffA + 4);
        ar[2] = ldg4(xrow + kc + koffA + 16);
        ar[3] = ldg4(xrow + kc + koffA + 20);
        float4 br[4][4];
        #pragma unroll
        for (int dt = 0; dt < 4; ++dt) {
            #pragma unroll
            for (int i = 0; i < 4; ++i)
                br[dt][i] = ldg4(wr[dt] + kc + koffB + 4 * i + (i >= 2 ? 8 : 0));
        }
        // ---- convert ----
        v16bf A;
        cvt8(A, 0, ar[0], ar[1]);
        cvt8(A, 8, ar[2], ar[3]);
        v16bf Bv[4];
        #pragma unroll
        for (int dt = 0; dt < 4; ++dt) {
            cvt8(Bv[dt], 0, br[dt][0], br[dt][1]);
            cvt8(Bv[dt], 8, br[dt][2], br[dt][3]);
        }
        // ---- 4 back-to-back WMMAs (shared A, disjoint D: no hazards) ----
        #pragma unroll
        for (int dt = 0; dt < 4; ++dt)
            acc[dt] = wmma_bf16(A, Bv[dt], acc[dt]);
    }

    // Store swizzled bf16 v. This wave's 16 tokens are one half of a 32-wide
    // k-chunk of the attention kernel.
    const int b    = mbase >> 11;              // /TSEQ
    const int tloc = mbase & (TSEQ - 1);
    const int kc_t = tloc >> 5;                // 32-token chunk
    const int half = (tloc >> 4) & 1;          // which 16 inside the chunk
    const int bh   = b * NHEAD + h;
    const int L2   = rlo + (hi ? 16 : 0);      // consumer lane

    #pragma unroll
    for (int dt = 0; dt < 4; ++dt) {
        const float bval = bias[h * HDIM + dt * 16 + rlo];
        Pack8 pk;
        #pragma unroll
        for (int r = 0; r < 8; ++r)
            pk.h[r] = (bf16_t)(acc[dt][r] + bval);
        const size_t off = (((size_t)bh * (TSEQ / 32) + kc_t) * 4 + dt) * 512
                         + (size_t)L2 * 16 + (half ? 8 : 0);
        *reinterpret_cast<uint4*>(vws + off) = pk.u;
    }
}

// ---------------------------------------------------------------------------
// Kernel 2: fused causal softmax + P@V. One wave owns 16 queries, streams the
// lower-triangular part of att exactly once (online softmax), accumulates
// O (16x64) in four f32 WMMA accumulators. Per-row stats live in lane
// (row%16); cross-half reduction is a lane^16 shuffle. Accumulator rescale
// (8 ds_bpermute + 32 v_mul) only runs when some row max actually increased
// (wave-uniform ballot branch) -- ~log(T) times per wave instead of every
// chunk, keeping the steady-state loop close to pure att streaming.
// ---------------------------------------------------------------------------
__global__ void __launch_bounds__(128) k_attn(const float* __restrict__ att,
                                              const bf16_t* __restrict__ vws,
                                              bf16_t* __restrict__ yws)
{
    const int lane  = threadIdx.x & 31;
    const int wid   = blockIdx.x * 4 + (threadIdx.x >> 5);   // 0..4095
    const int bh    = wid >> 7;                              // 0..31
    const int qt    = wid & 127;
    const int qbase = qt << 4;
    const int rlo   = lane & 15;
    const bool hi   = lane >= 16;
    const int q     = qbase + rlo;          // this lane's attention row
    const int koffA = hi ? 8 : 0;
    const float* arow = att + ((size_t)bh * TSEQ + q) * TSEQ;

    float m_run = -__builtin_inff();
    float s_run = 0.0f;
    v8f acc[4] = {};

    const int nchunk = (qbase + 15) / 32 + 1;   // causal: only k <= qmax
    for (int c = 0; c < nchunk; ++c) {
        const int k0 = c * 32;
        const float4 a0 = ldg4(arow + k0 + koffA);
        const float4 a1 = ldg4(arow + k0 + koffA + 4);
        const float4 a2 = ldg4(arow + k0 + koffA + 16);
        const float4 a3 = ldg4(arow + k0 + koffA + 20);
        if (c + 4 < nchunk)
            __builtin_prefetch(arow + k0 + 128, 0, 0);   // global_prefetch_b8

        float av[16] = { a0.x, a0.y, a0.z, a0.w, a1.x, a1.y, a1.z, a1.w,
                         a2.x, a2.y, a2.z, a2.w, a3.x, a3.y, a3.z, a3.w };

        // causal mask + zero-sentinel, chunk max
        float sc[16];
        float cm = -__builtin_inff();
        #pragma unroll
        for (int j = 0; j < 16; ++j) {
            const int kg = k0 + koffA + j + ((j >= 8) ? 8 : 0);
            const float a = av[j];
            const float vv = (a == 0.0f) ? -3.0f : a;
            sc[j] = (kg <= q) ? vv : -__builtin_inff();
            cm = fmaxf(cm, sc[j]);
        }
        cm = fmaxf(cm, __shfl_xor(cm, 16, 32));          // full-row max
        const float m_new = fmaxf(m_run, cm);            // finite after chunk 0
        const float alpha = __expf(m_run - m_new);

        float psum = 0.0f;
        v16bf Ae;
        #pragma unroll
        for (int j = 0; j < 16; ++j) {
            const float e = __expf(sc[j] - m_new);       // exp(-inf)=0 for masked
            psum += e;
            Ae[j] = (bf16_t)e;
        }
        const float rs = psum + __shfl_xor(psum, 16, 32);
        s_run = s_run * alpha + rs;
        m_run = m_new;

        // rescale O only if some row max increased (alpha<1 somewhere).
        // First chunk: alpha==0 but acc==0, skipping is still exact.
        if (__any(alpha < 1.0f)) {
            float f[8];
            #pragma unroll
            for (int r = 0; r < 8; ++r)
                f[r] = __shfl(alpha, r + (hi ? 8 : 0), 32);
            #pragma unroll
            for (int dt = 0; dt < 4; ++dt) {
                #pragma unroll
                for (int r = 0; r < 8; ++r)
                    acc[dt][r] *= f[r];
            }
        }

        // O += P (16x32 bf16) * V (32x16 bf16): B tiles preswizzled in vws.
        // Issue all 8 b128 loads, then 4 back-to-back WMMAs.
        const size_t tb = (((size_t)bh * (TSEQ / 32) + c) * 4) * 512
                        + (size_t)lane * 16;
        union { uint4 u[2]; v16bf v; } bu[4];
        #pragma unroll
        for (int dt = 0; dt < 4; ++dt) {
            const uint4* p = reinterpret_cast<const uint4*>(vws + tb + dt * 512);
            bu[dt].u[0] = p[0];
            bu[dt].u[1] = p[1];
        }
        #pragma unroll
        for (int dt = 0; dt < 4; ++dt)
            acc[dt] = wmma_bf16(Ae, bu[dt].v, acc[dt]);
    }

    // normalize by row sums and store y (bf16 row-major == WMMA-A pattern)
    const float inv = 1.0f / s_run;
    float fi[8];
    #pragma unroll
    for (int r = 0; r < 8; ++r)
        fi[r] = __shfl(inv, r + (hi ? 8 : 0), 32);

    const int b = bh >> 3, h = bh & 7;
    #pragma unroll
    for (int dt = 0; dt < 4; ++dt) {
        #pragma unroll
        for (int r = 0; r < 8; ++r) {
            const int t = qbase + r + (hi ? 8 : 0);
            yws[((size_t)b * TSEQ + t) * CDIM + h * HDIM + dt * 16 + rlo] =
                (bf16_t)(acc[dt][r] * fi[r]);
        }
    }
}

// ---------------------------------------------------------------------------
// Kernel 3: output projection. y is already bf16 row-major, which matches the
// 16-bit WMMA A layout (two contiguous 8-element runs per lane). Same
// loads-first restructure as K1.
// ---------------------------------------------------------------------------
__global__ void __launch_bounds__(128) k_oproj(const bf16_t* __restrict__ yws,
                                               const float* __restrict__ W,
                                               const float* __restrict__ bias,
                                               float* __restrict__ out)
{
    const int lane  = threadIdx.x & 31;
    const int wid   = blockIdx.x * 4 + (threadIdx.x >> 5);
    const int mtile = wid >> 3;
    const int nb    = wid & 7;
    const int mbase = mtile << 4;
    const int rlo   = lane & 15;
    const bool hi   = lane >= 16;
    const int koffA = hi ? 8 : 0;
    const int koffB = hi ? 8 : 0;

    const bf16_t* yrow = yws + (size_t)(mbase + rlo) * CDIM;
    const float* wr[4];
    #pragma unroll
    for (int dt = 0; dt < 4; ++dt)
        wr[dt] = W + (size_t)(nb * 64 + dt * 16 + rlo) * CDIM;

    v8f acc[4] = {};

    for (int kc = 0; kc < CDIM; kc += 32) {
        // ---- issue all raw loads first ----
        union { uint4 u[2]; v16bf v; } au;
        au.u[0] = *reinterpret_cast<const uint4*>(yrow + kc + koffA);
        au.u[1] = *reinterpret_cast<const uint4*>(yrow + kc + koffA + 16);
        float4 br[4][4];
        #pragma unroll
        for (int dt = 0; dt < 4; ++dt) {
            #pragma unroll
            for (int i = 0; i < 4; ++i)
                br[dt][i] = ldg4(wr[dt] + kc + koffB + 4 * i + (i >= 2 ? 8 : 0));
        }
        // ---- convert ----
        v16bf Bv[4];
        #pragma unroll
        for (int dt = 0; dt < 4; ++dt) {
            cvt8(Bv[dt], 0, br[dt][0], br[dt][1]);
            cvt8(Bv[dt], 8, br[dt][2], br[dt][3]);
        }
        // ---- WMMAs ----
        #pragma unroll
        for (int dt = 0; dt < 4; ++dt)
            acc[dt] = wmma_bf16(au.v, Bv[dt], acc[dt]);
    }

    #pragma unroll
    for (int dt = 0; dt < 4; ++dt) {
        const float bv = bias[nb * 64 + dt * 16 + rlo];
        #pragma unroll
        for (int r = 0; r < 8; ++r) {
            const int m = mbase + r + (hi ? 8 : 0);
            out[(size_t)m * CDIM + nb * 64 + dt * 16 + rlo] = acc[dt][r] + bv;
        }
    }
}

// ---------------------------------------------------------------------------
extern "C" void kernel_launch(void* const* d_in, const int* in_sizes, int n_in,
                              void* d_out, int out_size, void* d_ws, size_t ws_size,
                              hipStream_t stream)
{
    (void)in_sizes; (void)n_in; (void)out_size; (void)ws_size;

    const float* x      = (const float*)d_in[0];   // (4,2048,512)
    const float* att    = (const float*)d_in[1];   // (4,8,2048,2048)
    const float* W_attn = (const float*)d_in[2];   // (512,512)
    const float* b_attn = (const float*)d_in[3];   // (512)
    const float* W_proj = (const float*)d_in[4];   // (512,512)
    const float* b_proj = (const float*)d_in[5];   // (512)
    float* out = (float*)d_out;                    // (4,2048,512) f32

    // workspace: vws 8 MB (swizzled bf16 v) + yws 8 MB (bf16 y)
    bf16_t* vws = (bf16_t*)d_ws;
    bf16_t* yws = (bf16_t*)((char*)d_ws + (size_t)8 * 1024 * 1024);

    const dim3 blk(128);   // 4 waves (wave32)
    k_vproj<<<dim3(1024), blk, 0, stream>>>(x, W_attn, b_attn, vws);
    k_attn <<<dim3(1024), blk, 0, stream>>>(att, vws, yws);
    k_oproj<<<dim3(1024), blk, 0, stream>>>(yws, W_proj, b_proj, out);
}
